// Gallery_22204980920630
// MI455X (gfx1250) — matrix operand order repaired and checked by reading, not compiled, not run on hardware
//
#include <hip/hip_runtime.h>
#include <hip/hip_bf16.h>

typedef __attribute__((ext_vector_type(16))) __bf16 v16bf;
typedef __attribute__((ext_vector_type(8)))  float  v8f;

#define SIG_DIM  192
#define LDS_PITCH 200        // 400 B row stride = odd multiple of 16 B -> no
                             // LDS bank conflicts on per-lane ds_load_b128
#define BQ       128         // number of queries (fixed by problem)
#define TILE_N   64          // gallery columns per block
#define TOPK     32
#define KSTEPS   (SIG_DIM / 32)   // 6 WMMA k-steps

// ---------------------------------------------------------------------------
// Kernel 1: bf16x3 split-precision WMMA scoring.
// Each f32 input is split x = hi + lo (both bf16); q.g is computed as
// qh.gh + qh.gl + ql.gh on the bf16 WMMA pipe with f32 accumulation,
// giving ~1e-5 relative accuracy (the dropped ql.gl term is ~2^-16).
// Block = all 128 queries x 64 gallery columns; 256 threads = 8 waves,
// wave w owns query tile w, iterates 4 N-tiles x (6 k-steps x 3 WMMAs).
// Scores are written unmasked to ws; the mask is applied in the scan kernel.
// LDS rows are padded to 200 elements (odd multiple of 16 B) so the
// per-lane b128 fragment reads hit 16 distinct bank groups.
// ---------------------------------------------------------------------------
__global__ __launch_bounds__(256)
void score_kernel(const float* __restrict__ Q,
                  const float* __restrict__ G,
                  float* __restrict__ scores,
                  int N)
{
    __shared__ __bf16 qh[BQ * LDS_PITCH];        // 50 KB
    __shared__ __bf16 ql[BQ * LDS_PITCH];        // 50 KB
    __shared__ __bf16 gh[TILE_N * LDS_PITCH];    // 25 KB
    __shared__ __bf16 gl[TILE_N * LDS_PITCH];    // 25 KB   (150 KB total)

    const int tid    = threadIdx.x;
    const int n_base = blockIdx.x * TILE_N;

    // Split-convert queries (L2-resident; re-read per block is cheap).
    for (int i = tid; i < BQ * SIG_DIM; i += 256) {
        int m = i / SIG_DIM, k = i % SIG_DIM;
        float x  = Q[i];
        __bf16 h = (__bf16)x;
        qh[m * LDS_PITCH + k] = h;
        ql[m * LDS_PITCH + k] = (__bf16)(x - (float)h);
    }
    // Split-convert this block's gallery chunk.
    for (int i = tid; i < TILE_N * SIG_DIM; i += 256) {
        int n = i / SIG_DIM, k = i % SIG_DIM;
        int gn = n_base + n;
        float x = (gn < N) ? G[(size_t)gn * SIG_DIM + k] : 0.0f;
        __bf16 h = (__bf16)x;
        gh[n * LDS_PITCH + k] = h;
        gl[n * LDS_PITCH + k] = (__bf16)(x - (float)h);
    }
    __syncthreads();

    const int wave = tid >> 5;
    const int lane = tid & 31;
    const int grp  = lane >> 4;   // lane-group (ISA K-split)
    const int ln16 = lane & 15;
    const int m0   = wave * 16;   // this wave's query-tile base

    // Preload all A fragments (hi+lo) into registers; reused by all N-tiles.
    v16bf Ah[KSTEPS], Al[KSTEPS];
    #pragma unroll
    for (int kt = 0; kt < KSTEPS; ++kt) {
        const int k0 = kt * 32;
        #pragma unroll
        for (int h = 0; h < 16; ++h) {
            // A fragment (16x32 bf16): half h -> K = grp*16 + h
            int ka = grp * 16 + h;
            Ah[kt][h] = qh[(m0 + ln16) * LDS_PITCH + k0 + ka];
            Al[kt][h] = ql[(m0 + ln16) * LDS_PITCH + k0 + ka];
        }
    }

    for (int nt = 0; nt < TILE_N / 16; ++nt) {
        const int n0 = nt * 16;
        v8f acc = {};
        #pragma unroll
        for (int kt = 0; kt < KSTEPS; ++kt) {
            const int k0 = kt * 32;
            v16bf Bh, Bl;
            // B fragment (32x16 bf16): column n0+ln16, K = grp*16 + h.
            #pragma unroll
            for (int h = 0; h < 16; ++h) {
                int off = (n0 + ln16) * LDS_PITCH + k0 + grp * 16 + h;
                Bh[h] = gh[off];
                Bl[h] = gl[off];
            }
            // Small cross-terms first, dominant term last.
            acc = __builtin_amdgcn_wmma_f32_16x16x32_bf16(
                      false, Al[kt], false, Bh, (short)0, acc, false, false);
            acc = __builtin_amdgcn_wmma_f32_16x16x32_bf16(
                      false, Ah[kt], false, Bl, (short)0, acc, false, false);
            acc = __builtin_amdgcn_wmma_f32_16x16x32_bf16(
                      false, Ah[kt], false, Bh, (short)0, acc, false, false);
        }
        // C/D layout: VGPR r holds M = r + 8*grp, N = ln16.
        const int n = n_base + n0 + ln16;
        if (n < N) {
            #pragma unroll
            for (int r = 0; r < 8; ++r) {
                int m = m0 + r + 8 * grp;
                scores[(size_t)m * N + n] = acc[r];
            }
        }
    }
}

// ---------------------------------------------------------------------------
// Kernel 2: exact masked per-row top-32.  One block per query row.
// Each thread keeps a sorted (desc) local top-32 over its strided slice
// (skipping excluded entries), then 32 rounds of block-wide argmax over the
// thread heads select the global top-32.  Deterministic.
// ---------------------------------------------------------------------------
__global__ __launch_bounds__(256)
void topk_kernel(const float* __restrict__ scores,
                 const unsigned char* __restrict__ mask,
                 long long* __restrict__ idx_out,
                 float* __restrict__ val_out,
                 int N)
{
    const int b   = blockIdx.x;
    const int tid = threadIdx.x;
    const float*         row  = scores + (size_t)b * N;
    const unsigned char* mrow = mask   + (size_t)b * N;

    float v[TOPK];
    int   ix[TOPK];
    #pragma unroll
    for (int i = 0; i < TOPK; ++i) { v[i] = -__builtin_inff(); ix[i] = -1; }

    for (int n = tid; n < N; n += 256) {
        if (mrow[n]) continue;          // excluded -> acts as -inf
        float s = row[n];
        if (s > v[TOPK - 1]) {
            int pos = TOPK - 1;
            while (pos > 0 && v[pos - 1] < s) {
                v[pos] = v[pos - 1]; ix[pos] = ix[pos - 1]; --pos;
            }
            v[pos] = s; ix[pos] = n;
        }
    }

    __shared__ float sv[256];
    __shared__ int   si[256];
    __shared__ int   st[256];
    __shared__ int   winner;

    int head = 0;
    for (int k = 0; k < TOPK; ++k) {
        sv[tid] = (head < TOPK) ? v[head]  : -__builtin_inff();
        si[tid] = (head < TOPK) ? ix[head] : -1;
        st[tid] = tid;
        __syncthreads();
        for (int off = 128; off > 0; off >>= 1) {
            if (tid < off) {
                if (sv[tid + off] > sv[tid]) {
                    sv[tid] = sv[tid + off];
                    si[tid] = si[tid + off];
                    st[tid] = st[tid + off];
                }
            }
            __syncthreads();
        }
        if (tid == 0) {
            val_out[b * TOPK + k] = sv[0];
            idx_out[b * TOPK + k] = (long long)si[0];
            winner = st[0];
        }
        __syncthreads();
        if (tid == winner) ++head;
        __syncthreads();
    }
}

// ---------------------------------------------------------------------------
extern "C" void kernel_launch(void* const* d_in, const int* in_sizes, int n_in,
                              void* d_out, int out_size, void* d_ws, size_t ws_size,
                              hipStream_t stream)
{
    const float*         Qp   = (const float*)d_in[0];          // [B,192] f32
    const float*         Gp   = (const float*)d_in[1];          // [N,192] f32
    const unsigned char* mask = (const unsigned char*)d_in[2];  // [B,N] bool
    // d_in[3] = k (compile-time TOPK=32)

    const int N  = in_sizes[1] / SIG_DIM;
    const int Bq = in_sizes[0] / SIG_DIM;   // 128

    float*     ws_scores = (float*)d_ws;                        // [B,N] f32
    long long* idx_out   = (long long*)d_out;                   // [B,32] i64
    float*     val_out   = (float*)(idx_out + (size_t)Bq * TOPK);

    const int nblocks = (N + TILE_N - 1) / TILE_N;
    score_kernel<<<nblocks, 256, 0, stream>>>(Qp, Gp, ws_scores, N);
    topk_kernel<<<Bq, 256, 0, stream>>>(ws_scores, mask, idx_out, val_out, N);
}
